// GATLinkPrediction_14637248545240
// MI455X (gfx1250) — matrix operand order repaired and checked by reading, not compiled, not run on hardware
//
#include <hip/hip_runtime.h>
#include <hip/hip_bf16.h>
#include <math.h>

// Problem constants (match reference)
#define N_NODES 50000
#define N_EDGES 800000
#define E2      (N_EDGES + N_NODES)   // with self loops
#define DIM     128                   // all three GEMMs have Fin = 128
#define HEADS   4
#define CH      32
#define HC      (HEADS * CH)          // 128
#define NEG_SLOPE 0.2f

typedef __attribute__((ext_vector_type(16))) __bf16 v16bf;
typedef __attribute__((ext_vector_type(4)))  __bf16 v4bf;
typedef __attribute__((ext_vector_type(8)))  float  v8f;

// ---------------------------------------------------------------------------
// WMMA GEMM: Hout[n, FOUT] = A[n, 128] x W[128, FOUT]  (bf16 in, f32 acc).
// Block = 128 threads (4 waves). Block computes 64 rows x 16 cols; each wave
// one 16x16 tile, K fully unrolled (4 x wmma 16x16x32). FOUT is compile-time
// (128 or 32) so all strides are immediates.
//
// LDS is FRAGMENT-LINEAR: As[rowTile][kTile][lane] is the exact v16bf
// fragment lane `lane` feeds to v_wmma; consumer does one contiguous 32B
// LDS read per fragment (2x ds_load_b128), staging stores aligned 4-element
// bf16 quads (ds_store_b64) at permuted positions.
//
// A fragment mapping (16-bit A 16x32, wave32): lane = (row&15) + 16*khalf,
//   khalf=(c>>3)&1, i0(c quad-aligned) = c<8 ? c : (c<24 ? c-8 : c-16).
// B fragment mapping (16-bit B 32x16): lane = col + 16*((rr>>3)&1), i = rr<8 ? rr : (rr<24 ? rr-8 : rr-16).
// D: VGPR r -> row r + 8*khalf, col = lane&15.
// ---------------------------------------------------------------------------
template<int FOUT>
__global__ __launch_bounds__(128) void gemm_wmma_bf16(
    const float* __restrict__ A, const float* __restrict__ W,
    float* __restrict__ Hout, int n)
{
    constexpr int FIN = 128;
    constexpr int KT  = FIN / 32;          // 4 k-tiles
    __shared__ v16bf As[4][KT][32];        // 16 KB
    __shared__ v16bf Bs[KT][32];           // 4 KB

    const int t     = threadIdx.x;
    const int wv    = t >> 5;
    const int lane  = t & 31;
    const int lrow  = lane & 15;
    const int khalf = lane >> 4;
    const int m0    = blockIdx.x * 64;
    const int n0    = blockIdx.y * 16;
    const bool full = (m0 + 64 <= n);

    // ---- stage A panel: 64 rows x 128 cols, float4 per thread-iteration ----
    if (full) {
        #pragma unroll
        for (int q = 0; q < 16; ++q) {
            int qi = t + 128 * q;
            int r  = qi >> 5;
            int c4 = (qi & 31) * 4;
            float4 v = *(const float4*)&A[(size_t)(m0 + r) * FIN + c4];
            int kt = c4 >> 5, cc = c4 & 31;
            int kh = (cc >> 3) & 1;
            int i0 = (cc < 8) ? cc : (cc < 24 ? cc - 8 : cc - 16);
            v4bf p;
            p[0] = (__bf16)v.x; p[1] = (__bf16)v.y;
            p[2] = (__bf16)v.z; p[3] = (__bf16)v.w;
            *(v4bf*)(((__bf16*)&As[r >> 4][kt][(r & 15) + 16 * kh]) + i0) = p;
        }
    } else {
        #pragma unroll
        for (int q = 0; q < 16; ++q) {
            int qi = t + 128 * q;
            int r  = qi >> 5;
            int c4 = (qi & 31) * 4;
            float4 v = make_float4(0.f, 0.f, 0.f, 0.f);
            if (m0 + r < n) v = *(const float4*)&A[(size_t)(m0 + r) * FIN + c4];
            int kt = c4 >> 5, cc = c4 & 31;
            int kh = (cc >> 3) & 1;
            int i0 = (cc < 8) ? cc : (cc < 24 ? cc - 8 : cc - 16);
            v4bf p;
            p[0] = (__bf16)v.x; p[1] = (__bf16)v.y;
            p[2] = (__bf16)v.z; p[3] = (__bf16)v.w;
            *(v4bf*)(((__bf16*)&As[r >> 4][kt][(r & 15) + 16 * kh]) + i0) = p;
        }
    }

    // ---- stage B panel: 128 rows x 16 cols; 4 K-rows per thread-iteration ---
    {
        const int c  = t & 15;
        const int rq = t >> 4;             // 0..7
        const float* wp = &W[(size_t)n0 + c];
        #pragma unroll
        for (int q = 0; q < 4; ++q) {
            int r0 = (rq + 8 * q) * 4;     // 0..124, multiple of 4
            v4bf p;
            #pragma unroll
            for (int j = 0; j < 4; ++j)
                p[j] = (__bf16)wp[(size_t)(r0 + j) * FOUT];
            int kt = r0 >> 5, rr = r0 & 31;
            int kh = (rr >> 3) & 1, i0 = (rr < 8) ? rr : (rr < 24 ? rr - 8 : rr - 16);
            *(v4bf*)(((__bf16*)&Bs[kt][c + 16 * kh]) + i0) = p;
        }
    }
    __syncthreads();

    // ---- 4 x WMMA, fragments are contiguous 32B LDS reads ----
    v8f acc = {};
    #pragma unroll
    for (int kt = 0; kt < KT; ++kt) {
        v16bf af = As[wv][kt][lane];
        v16bf bf = Bs[kt][lane];
        acc = __builtin_amdgcn_wmma_f32_16x16x32_bf16(
            false, af, false, bf, (short)0, acc, false, false);
    }

    // ---- write back D: 8 rows, stride FOUT (compile-time) ----
    const int mbase = m0 + wv * 16 + 8 * khalf;
    float* outp = &Hout[(size_t)mbase * FOUT + n0 + lrow];
    if (full) {
        #pragma unroll
        for (int r = 0; r < 8; ++r)
            outp[(size_t)r * FOUT] = acc[r];
    } else {
        #pragma unroll
        for (int r = 0; r < 8; ++r)
            if (mbase + r < n) outp[(size_t)r * FOUT] = acc[r];
    }
}

// ---------------------------------------------------------------------------
// Per-node attention coefficients: asrc[n,h] = dot(H[n,h,:], a_src[h,:]).
// One wave per node; 32 lanes = 32 channels; shuffle reduction.
// ---------------------------------------------------------------------------
__global__ __launch_bounds__(256) void alphas_kernel(
    const float* __restrict__ Hf, const float* __restrict__ avS,
    const float* __restrict__ avD, float* __restrict__ asrc,
    float* __restrict__ adst, int n, int heads)
{
    int wid  = (int)((blockIdx.x * blockDim.x + threadIdx.x) >> 5);
    int lane = threadIdx.x & 31;
    if (wid >= n) return;
    int F = heads * 32;
    for (int h = 0; h < heads; ++h) {
        float hv = Hf[(size_t)wid * F + h * 32 + lane];
        float s = hv * avS[h * 32 + lane];
        float d = hv * avD[h * 32 + lane];
        #pragma unroll
        for (int off = 16; off; off >>= 1) {
            s += __shfl_xor(s, off, 32);
            d += __shfl_xor(d, off, 32);
        }
        if (lane == 0) {
            asrc[(size_t)wid * heads + h] = s;
            adst[(size_t)wid * heads + h] = d;
        }
    }
}

// monotone float <-> uint key (atomic max over signed floats)
__device__ __forceinline__ unsigned fkey(float f) {
    unsigned u = __float_as_uint(f);
    return (u & 0x80000000u) ? ~u : (u | 0x80000000u);
}
__device__ __forceinline__ float funkey(unsigned k) {
    unsigned u = (k & 0x80000000u) ? (k & 0x7fffffffu) : ~k;
    return __uint_as_float(u);
}

__device__ __forceinline__ void edge_sd(int e, const int* __restrict__ src,
                                        const int* __restrict__ dst, int& s, int& d) {
    if (e < N_EDGES) { s = src[e]; d = dst[e]; }
    else             { s = e - N_EDGES; d = s; }   // self loops
}

__device__ __forceinline__ float leaky(float x) {
    return x > 0.0f ? x : NEG_SLOPE * x;
}

// Pass 1: segment max of leaky-relu logits
__global__ __launch_bounds__(256) void edge_max_kernel(
    const int* __restrict__ src, const int* __restrict__ dst,
    const float* __restrict__ asrc, const float* __restrict__ adst,
    unsigned* __restrict__ mkey, int heads)
{
    int e = blockIdx.x * blockDim.x + threadIdx.x;
    if (e >= E2) return;
    int s, d; edge_sd(e, src, dst, s, d);
    for (int h = 0; h < heads; ++h) {
        float x = leaky(asrc[(size_t)s * heads + h] + adst[(size_t)d * heads + h]);
        atomicMax(&mkey[(size_t)d * heads + h], fkey(x));
    }
}

// Pass 2: segment sum of exp(logit - max)
__global__ __launch_bounds__(256) void edge_expsum_kernel(
    const int* __restrict__ src, const int* __restrict__ dst,
    const float* __restrict__ asrc, const float* __restrict__ adst,
    const unsigned* __restrict__ mkey, float* __restrict__ denom, int heads)
{
    int e = blockIdx.x * blockDim.x + threadIdx.x;
    if (e >= E2) return;
    int s, d; edge_sd(e, src, dst, s, d);
    for (int h = 0; h < heads; ++h) {
        float x = leaky(asrc[(size_t)s * heads + h] + adst[(size_t)d * heads + h]);
        float m = funkey(mkey[(size_t)d * heads + h]);
        atomicAdd(&denom[(size_t)d * heads + h], expf(x - m));
    }
}

// Pass 3: weighted message scatter. One wave per edge; 32 lanes = channels.
__global__ __launch_bounds__(256) void edge_msg_kernel(
    const int* __restrict__ src, const int* __restrict__ dst,
    const float* __restrict__ Hf, const float* __restrict__ asrc,
    const float* __restrict__ adst, const unsigned* __restrict__ mkey,
    const float* __restrict__ denom, float* __restrict__ out, int heads)
{
    int wid  = (int)((blockIdx.x * blockDim.x + threadIdx.x) >> 5);
    int lane = threadIdx.x & 31;
    if (wid >= E2) return;
    int s, d; edge_sd(wid, src, dst, s, d);
    int F = heads * 32;
    for (int h = 0; h < heads; ++h) {
        float x  = leaky(asrc[(size_t)s * heads + h] + adst[(size_t)d * heads + h]);
        float m  = funkey(mkey[(size_t)d * heads + h]);
        float al = expf(x - m) / (denom[(size_t)d * heads + h] + 1e-16f);
        float v  = Hf[(size_t)s * F + h * 32 + lane] * al;
        atomicAdd(&out[(size_t)d * F + h * 32 + lane], v);
    }
}

// Epilogue: out += bias; optional ELU. F is a power of two (128 or 32).
__global__ __launch_bounds__(256) void bias_act_kernel(
    float* __restrict__ out, const float* __restrict__ b, int n, int fmask, int doElu)
{
    size_t i = (size_t)blockIdx.x * blockDim.x + threadIdx.x;
    if (i >= (size_t)n * (fmask + 1)) return;
    float v = out[i] + b[(int)i & fmask];
    if (doElu) v = v > 0.0f ? v : (expf(v) - 1.0f);
    out[i] = v;
}

// ---------------------------------------------------------------------------
extern "C" void kernel_launch(void* const* d_in, const int* in_sizes, int n_in,
                              void* d_out, int out_size, void* d_ws, size_t ws_size,
                              hipStream_t stream)
{
    const float* x   = (const float*)d_in[0];
    const int*   ei  = (const int*)  d_in[1];
    const int*   src = ei;
    const int*   dst = ei + N_EDGES;
    const float* W1  = (const float*)d_in[2];
    const float* aS1 = (const float*)d_in[3];
    const float* aD1 = (const float*)d_in[4];
    const float* b1  = (const float*)d_in[5];
    const float* W2  = (const float*)d_in[6];
    const float* aS2 = (const float*)d_in[7];
    const float* aD2 = (const float*)d_in[8];
    const float* b2  = (const float*)d_in[9];
    const float* W3  = (const float*)d_in[10];
    const float* aS3 = (const float*)d_in[11];
    const float* aD3 = (const float*)d_in[12];
    const float* b3  = (const float*)d_in[13];

    // workspace layout
    float*    Hbuf  = (float*)d_ws;                          // N x 128
    float*    Obuf  = Hbuf + (size_t)N_NODES * HC;           // N x 128
    float*    asrc  = Obuf + (size_t)N_NODES * HC;           // N x HEADS
    float*    adst  = asrc + (size_t)N_NODES * HEADS;        // N x HEADS
    unsigned* mkey  = (unsigned*)(adst + (size_t)N_NODES * HEADS);
    float*    denom = (float*)(mkey + (size_t)N_NODES * HEADS);

    const int edgeBlocks     = (E2 + 255) / 256;       // thread-per-edge kernels
    const int edgeWaveBlocks = (E2 + 7) / 8;           // wave-per-edge kernels
    const int nodeWaveBlocks = (N_NODES + 7) / 8;

    auto run_attention = [&](const float* aS, const float* aD, const float* bias,
                             int heads, float* outbuf, int doElu) {
        const int fout = heads * 32;
        // per-node attention dot products
        hipLaunchKernelGGL(alphas_kernel, dim3(nodeWaveBlocks), dim3(256), 0, stream,
                           Hbuf, aS, aD, asrc, adst, N_NODES, heads);
        // zero segment-softmax state + output accumulator
        hipMemsetAsync(mkey,  0, (size_t)N_NODES * heads * sizeof(unsigned), stream);
        hipMemsetAsync(denom, 0, (size_t)N_NODES * heads * sizeof(float), stream);
        hipMemsetAsync(outbuf, 0, (size_t)N_NODES * fout * sizeof(float), stream);
        // segment softmax (3 passes) + message aggregation
        hipLaunchKernelGGL(edge_max_kernel, dim3(edgeBlocks), dim3(256), 0, stream,
                           src, dst, asrc, adst, mkey, heads);
        hipLaunchKernelGGL(edge_expsum_kernel, dim3(edgeBlocks), dim3(256), 0, stream,
                           src, dst, asrc, adst, mkey, denom, heads);
        hipLaunchKernelGGL(edge_msg_kernel, dim3(edgeWaveBlocks), dim3(256), 0, stream,
                           src, dst, Hbuf, asrc, adst, mkey, denom, outbuf, heads);
        // bias (+ ELU)
        size_t tot = (size_t)N_NODES * fout;
        hipLaunchKernelGGL(bias_act_kernel, dim3((unsigned)((tot + 255) / 256)),
                           dim3(256), 0, stream, outbuf, bias, N_NODES, fout - 1, doElu);
    };

    const dim3 g128((N_NODES + 63) / 64, HC / 16);   // 782 x 8
    const dim3 g32 ((N_NODES + 63) / 64, CH / 16);   // 782 x 2

    // Layer 1: x(128) -> 4 heads x 32, concat, ELU    (out in Obuf)
    hipLaunchKernelGGL(HIP_KERNEL_NAME(gemm_wmma_bf16<HC>), g128, dim3(128), 0, stream,
                       x, W1, Hbuf, N_NODES);
    run_attention(aS1, aD1, b1, HEADS, Obuf, 1);

    // Layer 2: Obuf(128) -> 4x32, concat, ELU  (Obuf re-zeroed only after the
    //          GEMM consumed it — stream order guarantees this)
    hipLaunchKernelGGL(HIP_KERNEL_NAME(gemm_wmma_bf16<HC>), g128, dim3(128), 0, stream,
                       Obuf, W2, Hbuf, N_NODES);
    run_attention(aS2, aD2, b2, HEADS, Obuf, 1);

    // Layer 3: Obuf(128) -> 1 head x 32, mean==identity, +b3, no ELU -> d_out
    hipLaunchKernelGGL(HIP_KERNEL_NAME(gemm_wmma_bf16<CH>), g32, dim3(128), 0, stream,
                       Obuf, W3, Hbuf, N_NODES);
    run_attention(aS3, aD3, b3, 1, (float*)d_out, 0);
}
